// Qwen3_5Attention_39874476376654
// MI455X (gfx1250) — matrix operand order repaired and checked by reading, not compiled, not run on hardware
//
#include <hip/hip_runtime.h>
#include <hip/hip_bf16.h>
#include <cmath>

// ---------------------------------------------------------------------------
// Types for WMMA fragments (CDNA5 gfx1250, wave32)
// ---------------------------------------------------------------------------
typedef __attribute__((ext_vector_type(16))) __bf16 v16bf;
typedef __attribute__((ext_vector_type(8)))  float  v8f;

union FragU {
    v16bf    v;
    unsigned u[8];
    uint4    q[2];
};

// Split a pair of f32 into truncated-bf16 hi pack and bf16(lo) pack.
// x = hi + lo exactly in f32; bf16(lo) captures the next 8 mantissa bits.
__device__ __forceinline__ void pack2(float e0, float e1, unsigned& h, unsigned& l)
{
    unsigned u0 = __float_as_uint(e0), u1 = __float_as_uint(e1);
    unsigned h0 = u0 & 0xFFFF0000u,    h1 = u1 & 0xFFFF0000u;
    h = h1 | (h0 >> 16);
    float l0 = e0 - __uint_as_float(h0);
    float l1 = e1 - __uint_as_float(h1);
    l = (__float_as_uint(l1) & 0xFFFF0000u) | (__float_as_uint(l0) >> 16);
}

#define PA 36   // LDS pitch in dwords (pair-index space), 16B aligned, bank-staggered
#define PB 36

// ---------------------------------------------------------------------------
// Split-bf16 GEMM:  out[m,n] = sum_k A[m,k] * W[n,k]   (A: MxK, W: NxK)
// Block tile 64M x 128N, 8 waves (2x4), wave tile 32x32, BK=64 LDS rounds.
// A/B panels are split-converted to bf16 hi/lo ONCE per block into LDS;
// waves fetch fragments with ds_load_b128.
// MODE 0: + bias, fused depthwise-conv epilogue  -> core_base
// MODE 1: sigmoid epilogue                       -> gate
// MODE 2: plain store                            -> d_out
// ---------------------------------------------------------------------------
template <int MODE>
__global__ __launch_bounds__(256)
void gemm_split_bf16(const float* __restrict__ A,
                     const float* __restrict__ W,
                     int K, int N,
                     const float* __restrict__ bias,       // MODE 0 (b_qkv+4096)
                     const float* __restrict__ convk,      // MODE 0 (conv_kernel+4096*4)
                     const float* __restrict__ conv_state, // MODE 0 (B,4,8192)
                     float* __restrict__ out)
{
    __shared__ unsigned sAhi[64 * PA];
    __shared__ unsigned sAlo[64 * PA];
    __shared__ unsigned sBhi[128 * PB];
    __shared__ unsigned sBlo[128 * PB];

    const int tid  = threadIdx.x;
    const int wave = tid >> 5;
    const int lane = tid & 31;
    const int half = lane >> 4;     // 0: lanes 0-15, 1: lanes 16-31
    const int r    = lane & 15;
    const int wm   = wave >> 2;     // 0..1 : 32-row slab within block
    const int wn   = wave & 3;      // 0..3 : 32-col slab within block

    const int blockM = blockIdx.x * 64;
    const int blockN = blockIdx.y * 128;

    v8f acc[2][2] = {{{}, {}}, {{}, {}}};

    for (int k0 = 0; k0 < K; k0 += 64) {
        if (k0) __syncthreads();

        // ---- Stage + split-convert A panel: 64 rows x 64 k (2048 f32 pairs) ----
        #pragma unroll
        for (int i = 0; i < 8; ++i) {
            const int p   = tid + 256 * i;
            const int row = p >> 5;
            const int pc  = p & 31;
            const float2 e = *(const float2*)(A + (size_t)(blockM + row) * K + k0 + 2 * pc);
            if (i == 0 && k0 + 64 < K)
                __builtin_prefetch(A + (size_t)(blockM + row) * K + k0 + 64 + 2 * pc, 0, 1);
            pack2(e.x, e.y, sAhi[row * PA + pc], sAlo[row * PA + pc]);
        }
        // ---- Stage + split-convert B panel: 128 cols x 64 k (4096 f32 pairs) ----
        #pragma unroll
        for (int i = 0; i < 16; ++i) {
            const int p   = tid + 256 * i;
            const int row = p >> 5;
            const int pc  = p & 31;
            const float2 e = *(const float2*)(W + (size_t)(blockN + row) * K + k0 + 2 * pc);
            if (i == 0 && k0 + 64 < K)
                __builtin_prefetch(W + (size_t)(blockN + row) * K + k0 + 64 + 2 * pc, 0, 1);
            pack2(e.x, e.y, sBhi[row * PB + pc], sBlo[row * PB + pc]);
        }
        __syncthreads();

        // ---- Two K=32 WMMA substeps out of the 64-wide LDS panel ----
        #pragma unroll
        for (int ks = 0; ks < 2; ++ks) {
            const int pb = ks * 16;   // pair-index base of this substep

            FragU ahi[2], alo[2], bhi[2], blo[2];
            #pragma unroll
            for (int mt = 0; mt < 2; ++mt) {
                // A lane layout: pairs [pb+4h .. +3] -> u0..3 ; [pb+8+4h .. +3] -> u4..7
                const int base = (wm * 32 + mt * 16 + r) * PA + pb + 4 * half;
                ahi[mt].q[0] = *(const uint4*)&sAhi[base];
                ahi[mt].q[1] = *(const uint4*)&sAhi[base + 8];
                alo[mt].q[0] = *(const uint4*)&sAlo[base];
                alo[mt].q[1] = *(const uint4*)&sAlo[base + 8];
            }
            #pragma unroll
            for (int nt = 0; nt < 2; ++nt) {
                // B lane layout: pairs [pb+4h .. +3] -> u0..3 ; [pb+8+4h .. +3] -> u4..7
                const int base = (wn * 32 + nt * 16 + r) * PB + pb + 4 * half;
                bhi[nt].q[0] = *(const uint4*)&sBhi[base];
                bhi[nt].q[1] = *(const uint4*)&sBhi[base + 8];
                blo[nt].q[0] = *(const uint4*)&sBlo[base];
                blo[nt].q[1] = *(const uint4*)&sBlo[base + 8];
            }

            // 3-term compensated product per tile: lo*hi + hi*lo + hi*hi
            #pragma unroll
            for (int mt = 0; mt < 2; ++mt) {
                #pragma unroll
                for (int nt = 0; nt < 2; ++nt) {
                    acc[mt][nt] = __builtin_amdgcn_wmma_f32_16x16x32_bf16(
                        false, alo[mt].v, false, bhi[nt].v, (short)0, acc[mt][nt], false, false);
                    acc[mt][nt] = __builtin_amdgcn_wmma_f32_16x16x32_bf16(
                        false, ahi[mt].v, false, blo[nt].v, (short)0, acc[mt][nt], false, false);
                    acc[mt][nt] = __builtin_amdgcn_wmma_f32_16x16x32_bf16(
                        false, ahi[mt].v, false, bhi[nt].v, (short)0, acc[mt][nt], false, false);
                }
            }
        }
    }

    // ---- Epilogue: C/D layout -> (row = base + v + 8*half, col = base + r) ----
    #pragma unroll
    for (int mt = 0; mt < 2; ++mt) {
        #pragma unroll
        for (int nt = 0; nt < 2; ++nt) {
            #pragma unroll
            for (int v = 0; v < 8; ++v) {
                const int row = blockM + wm * 32 + mt * 16 + half * 8 + v;
                const int col = blockN + wn * 32 + nt * 16 + r;
                float val = acc[mt][nt][v];
                if (MODE == 0) {
                    // mixed_qkv[row, 4096+col] + causal conv with conv_state planes 1..3
                    val += bias[col];
                    const float4 ck = *(const float4*)(convk + (size_t)col * 4);
                    const float* cs = conv_state + (size_t)row * 32768 + 4096 + col;
                    float o = cs[8192] * ck.x + cs[16384] * ck.y + cs[24576] * ck.z + val * ck.w;
                    out[(size_t)row * N + col] = o;
                } else if (MODE == 1) {
                    out[(size_t)row * N + col] = 1.0f / (1.0f + expf(-val));
                } else {
                    out[(size_t)row * N + col] = val;
                }
            }
        }
    }
}

// ---------------------------------------------------------------------------
// Alpha/beta projections: ab[b, c] = hidden[b] . W_alpha[c], ab[b, 32+c] = . W_beta[c]
// 8 blocks x 32 batch rows; weights staged through LDS so they are read once/block.
// ---------------------------------------------------------------------------
__global__ __launch_bounds__(256)
void alphabeta_kernel(const float* __restrict__ hidden,
                      const float* __restrict__ Wa,
                      const float* __restrict__ Wb,
                      float* __restrict__ ab)
{
    __shared__ float sh[32 * 64];
    __shared__ float swa[32 * 64];
    __shared__ float swb[32 * 64];

    const int b0  = blockIdx.x * 32;
    const int t   = threadIdx.x;
    const int c   = t & 31;
    const int grp = t >> 5;  // 0..7, each handles 4 rows

    float accA[4] = {0.f, 0.f, 0.f, 0.f};
    float accB[4] = {0.f, 0.f, 0.f, 0.f};

    for (int k0 = 0; k0 < 2048; k0 += 64) {
        __syncthreads();
        for (int i = t; i < 2048; i += 256) {
            const int rr = i >> 6, kk = i & 63;
            sh[i]  = hidden[(size_t)(b0 + rr) * 2048 + k0 + kk];
            swa[i] = Wa[(size_t)rr * 2048 + k0 + kk];
            swb[i] = Wb[(size_t)rr * 2048 + k0 + kk];
        }
        __syncthreads();
        for (int kk = 0; kk < 64; ++kk) {
            const float wa = swa[c * 64 + kk];
            const float wb = swb[c * 64 + kk];
            #pragma unroll
            for (int j = 0; j < 4; ++j) {
                const float h = sh[(grp * 4 + j) * 64 + kk];
                accA[j] = fmaf(h, wa, accA[j]);
                accB[j] = fmaf(h, wb, accB[j]);
            }
        }
    }
    #pragma unroll
    for (int j = 0; j < 4; ++j) {
        const int row = b0 + grp * 4 + j;
        ab[(size_t)row * 64 + c]      = accA[j];
        ab[(size_t)row * 64 + 32 + c] = accB[j];
    }
}

// ---------------------------------------------------------------------------
// Per-batch-row fusion: group means, SSM state update, repeat-expand,
// RMS-norm, gate multiply -> core_final
// ---------------------------------------------------------------------------
__global__ __launch_bounds__(256)
void fuse_kernel(const float* __restrict__ core_base,
                 const float* __restrict__ gate,
                 const float* __restrict__ ab,
                 const float* __restrict__ ssm_a,
                 const float* __restrict__ dt_bias,
                 const float* __restrict__ ssm_state,
                 const float* __restrict__ norm_w,
                 float* __restrict__ core_final)
{
    const int b = blockIdx.x;
    const int t = threadIdx.x;

    __shared__ float skg[32], svg[32], ssig[32];
    __shared__ float ssum;
    __shared__ float smw;

    if (t < 32) { skg[t] = 0.f; svg[t] = 0.f; }
    if (t == 0) {
        ssum = 0.f;
        float s = 0.f;
        for (int i = 0; i < 128; ++i) s += norm_w[i];
        smw = s * (1.0f / 128.0f);
    }
    __syncthreads();

    const float* cb = core_base + (size_t)b * 4096;
    for (int i = 0; i < 16; ++i) {
        const int j = t + 256 * i;
        const float v = cb[j];
        if (j < 2048) atomicAdd(&skg[j >> 6], v);                // k groups (width 64)
        else          atomicAdd(&svg[(j - 2048) >> 6], v);       // v groups
    }
    __syncthreads();

    if (t < 32) {
        const float av = ab[(size_t)b * 64 + t];
        const float bv = ab[(size_t)b * 64 + 32 + t];
        const float kg = skg[t] * (1.0f / 64.0f);
        const float vg = svg[t] * (1.0f / 64.0f);
        const float x  = av + dt_bias[t];
        const float sp = (x > 20.0f) ? x : log1pf(expf(x));      // softplus
        const float g  = -expf(ssm_a[t]) * sp;
        const float bt = 1.0f / (1.0f + expf(-bv));              // beta
        const float ns = expf(g) * ssm_state[(size_t)b * 32 + t] + bt * vg;
        ssig[t] = ns * (1.0f / (1.0f + expf(-kg)));              // state_signal
    }
    __syncthreads();

    float* cf = core_final + (size_t)b * 4096;
    float ss = 0.f;
    for (int i = 0; i < 16; ++i) {
        const int j = t + 256 * i;
        const float v = cb[j] + ssig[j >> 7];                    // repeat_factor = 128
        cf[j] = v;
        ss += v * v;
    }
    atomicAdd(&ssum, ss);
    __syncthreads();

    const float rn = rsqrtf(ssum * (1.0f / 4096.0f) + 1e-6f);
    const float* gt = gate + (size_t)b * 4096;
    for (int i = 0; i < 16; ++i) {
        const int j = t + 256 * i;
        cf[j] = cf[j] * rn * gt[j] * smw;
    }
}

// ---------------------------------------------------------------------------
// Launch
// ---------------------------------------------------------------------------
extern "C" void kernel_launch(void* const* d_in, const int* in_sizes, int n_in,
                              void* d_out, int out_size, void* d_ws, size_t ws_size,
                              hipStream_t stream)
{
    (void)in_sizes; (void)n_in; (void)out_size; (void)ws_size;

    const float* hidden     = (const float*)d_in[0];   // (256, 1, 2048)
    const float* W_qkv      = (const float*)d_in[1];   // (8192, 2048)
    const float* b_qkv      = (const float*)d_in[2];   // (8192,)
    const float* W_gate     = (const float*)d_in[3];   // (4096, 2048)
    const float* W_alpha    = (const float*)d_in[4];   // (32, 2048)
    const float* W_beta     = (const float*)d_in[5];   // (32, 2048)
    const float* W_out      = (const float*)d_in[6];   // (2048, 4096)
    const float* ssm_a      = (const float*)d_in[7];   // (32,)
    const float* dt_bias    = (const float*)d_in[8];   // (32,)
    const float* norm_w     = (const float*)d_in[9];   // (128,)
    const float* conv_k     = (const float*)d_in[10];  // (8192, 4)
    const float* ssm_state  = (const float*)d_in[11];  // (256, 32)
    const float* conv_state = (const float*)d_in[12];  // (256, 4, 8192)
    float* out = (float*)d_out;                        // (256, 1, 2048) f32

    char* ws = (char*)d_ws;
    float* core_base  = (float*)ws;                                   // 4 MB
    float* gate       = (float*)(ws + ((size_t)4 << 20));             // 4 MB
    float* ab         = (float*)(ws + ((size_t)8 << 20));             // 64 KB
    float* core_final = (float*)(ws + ((size_t)8 << 20) + 65536);     // 4 MB

    const dim3 blk(256);

    // 1) core_base = conv( hidden @ W_qkv[4096:]^T + b_qkv[4096:] , conv_state )
    gemm_split_bf16<0><<<dim3(256 / 64, 4096 / 128), blk, 0, stream>>>(
        hidden, W_qkv + (size_t)4096 * 2048, 2048, 4096,
        b_qkv + 4096, conv_k + (size_t)4096 * 4, conv_state, core_base);

    // 2) gate = sigmoid( hidden @ W_gate^T )
    gemm_split_bf16<1><<<dim3(256 / 64, 4096 / 128), blk, 0, stream>>>(
        hidden, W_gate, 2048, 4096, nullptr, nullptr, nullptr, gate);

    // 3) alpha/beta projections
    alphabeta_kernel<<<dim3(8), blk, 0, stream>>>(hidden, W_alpha, W_beta, ab);

    // 4) SSM update + RMS-norm + gating
    fuse_kernel<<<dim3(256), blk, 0, stream>>>(
        core_base, gate, ab, ssm_a, dt_bias, ssm_state, norm_w, core_final);

    // 5) out = core_final @ W_out^T
    gemm_split_bf16<2><<<dim3(256 / 64, 2048 / 128), blk, 0, stream>>>(
        core_final, W_out, 4096, 2048, nullptr, nullptr, nullptr, out);
}
